// CausalSelfAttention_44899588112869
// MI455X (gfx1250) — matrix operand order repaired and checked by reading, not compiled, not run on hardware
//
#include <hip/hip_runtime.h>
#include <math.h>

typedef __attribute__((ext_vector_type(16))) __bf16 v16bf;
typedef __attribute__((ext_vector_type(8)))  float  v8f;

static constexpr int B_  = 2;
static constexpr int H_  = 16;
static constexpr int N_  = 2048;
static constexpr int D_  = 1024;
static constexpr int DH_ = 64;

// Load a 16-element bf16 fragment as two 128-bit LDS reads (ds_load_b128).
static __device__ __forceinline__ v16bf ldfrag(const __bf16* p0, const __bf16* p1) {
  union { uint4 q[2]; v16bf v; } u;
  u.q[0] = *(const uint4*)p0;
  u.q[1] = *(const uint4*)p1;
  return u.v;
}

// --- gfx1250 async global->LDS copy (ASYNCcnt path), guarded for portability ---
#if __has_builtin(__builtin_amdgcn_global_load_async_to_lds_b128)
#define ASYNC_LDS 1
#endif

typedef int vi4_b128 __attribute__((vector_size(16)));

static __device__ __forceinline__ void g2l_b128(void* lds, const void* g) {
#ifdef ASYNC_LDS
  __builtin_amdgcn_global_load_async_to_lds_b128(
      (__attribute__((address_space(1))) vi4_b128*)(g),
      (__attribute__((address_space(3))) vi4_b128*)(lds), 0, 0);
#else
  *(uint4*)lds = *(const uint4*)g;
#endif
}

static __device__ __forceinline__ void async_wait() {
#ifdef ASYNC_LDS
#if __has_builtin(__builtin_amdgcn_s_wait_asynccnt)
  __builtin_amdgcn_s_wait_asynccnt(0);
#else
  asm volatile("s_wait_asynccnt 0" ::: "memory");
#endif
#endif
}

// ---------------------------------------------------------------------------
// Kernel 1: per-head QKV projection.
//   qkv[b,h,n,e] = sum_d x[b,n,d]*W_qkv[h,d,e] + b_qkv[h,e],  e in [0,192)
//   split order (reference): e<64 -> K, 64<=e<128 -> Q (x1/8 folded), else V
//   K,Q stored bf16 [b][h][n][64];  V stored TRANSPOSED bf16 [b][h][dh][n]
// ---------------------------------------------------------------------------
__global__ __launch_bounds__(256) void qkv_kernel(
    const float* __restrict__ x, const float* __restrict__ Wqkv,
    const float* __restrict__ bqkv,
    __bf16* __restrict__ kws, __bf16* __restrict__ qws, __bf16* __restrict__ vws) {
  __shared__ __align__(16) __bf16 lx[64][40];     // x tile, row-major [n][d]
  __shared__ __align__(16) __bf16 lwT[192][40];   // W tile transposed [e][d]
  const int nblk = blockIdx.x, h = blockIdx.y, b = blockIdx.z;
  const int tid = threadIdx.x, lane = tid & 31, wv = tid >> 5;
  const int mtile = wv & 3;          // 4 M-tiles of 16 rows
  const int nt0   = (wv >> 2) * 6;   // 12 N-tiles of 16 cols, 6 per wave-half
  v8f acc[6] = {};

  const float* xblk = x + ((size_t)b * N_ + (size_t)nblk * 64) * D_;
  const float* wblk = Wqkv + (size_t)h * D_ * 192;

  for (int kc = 0; kc < D_ / 32; ++kc) {
    // stage x tile 64x32: f32 -> bf16, packed 128-bit LDS store
    {
      int idx = tid * 8;
      int r = idx >> 5, c = idx & 31;
      const float4* s = (const float4*)(xblk + (size_t)r * D_ + kc * 32 + c);
      float4 f0 = s[0], f1 = s[1];
      union { __bf16 h[8]; uint4 q; } pk;
      pk.h[0]=(__bf16)f0.x; pk.h[1]=(__bf16)f0.y; pk.h[2]=(__bf16)f0.z; pk.h[3]=(__bf16)f0.w;
      pk.h[4]=(__bf16)f1.x; pk.h[5]=(__bf16)f1.y; pk.h[6]=(__bf16)f1.z; pk.h[7]=(__bf16)f1.w;
      *(uint4*)&lx[r][c] = pk.q;
    }
    // stage W tile 32x192 transposed -> lwT[e][d]
    {
      int idx = tid * 24;
      int r = idx / 192, c = idx % 192;       // r: d-row, c: e-col
      const float* s = wblk + (size_t)(kc * 32 + r) * 192 + c;
      for (int i = 0; i < 24; i += 4) {
        float4 f = *(const float4*)(s + i);
        lwT[c + i    ][r] = (__bf16)f.x;
        lwT[c + i + 1][r] = (__bf16)f.y;
        lwT[c + i + 2][r] = (__bf16)f.z;
        lwT[c + i + 3][r] = (__bf16)f.w;
      }
    }
    if (kc + 1 < D_ / 32) {
      __builtin_prefetch(xblk + (size_t)(tid >> 2) * D_ + (kc + 1) * 32, 0, 3);
      __builtin_prefetch(wblk + (size_t)((kc + 1) * 32 + (tid & 31)) * 192, 0, 3);
    }
    __syncthreads();

    // preload all fragments, then run the WMMA chain
    const int row = mtile * 16 + (lane & 15);
    const int hb  = (lane >> 4) * 8;
    const int kb  = (lane >> 4) * 16;
    v16bf a = ldfrag(&lx[row][hb], &lx[row][16 + hb]);
    v16bf bm[6];
    for (int t = 0; t < 6; ++t) {
      int col = (nt0 + t) * 16 + (lane & 15);
      bm[t] = ldfrag(&lwT[col][hb], &lwT[col][16 + hb]);
    }
    for (int t = 0; t < 6; ++t)
      acc[t] = __builtin_amdgcn_wmma_f32_16x16x32_bf16(
          false, a, false, bm[t], (short)0, acc[t], false, false);
    __syncthreads();
  }

  // epilogue: bias, route to k/q/v (q scaled by 1/8, v transposed)
  for (int t = 0; t < 6; ++t) {
    int e = (nt0 + t) * 16 + (lane & 15);
    int region = e >> 6;             // 0=k, 1=q, 2=v (reference split order)
    int dh = e & 63;
    float bias = bqkv[h * 192 + e];
    size_t base = ((size_t)b * H_ + h) * (size_t)N_ * DH_;
    for (int r = 0; r < 8; ++r) {
      int row = mtile * 16 + r + ((lane >> 4) * 8);
      int ng = nblk * 64 + row;
      float v = acc[t][r] + bias;
      if (region == 0) {
        kws[base + (size_t)ng * DH_ + dh] = (__bf16)v;
      } else if (region == 1) {
        qws[base + (size_t)ng * DH_ + dh] = (__bf16)(v * 0.125f);
      } else {
        vws[base + (size_t)dh * N_ + ng] = (__bf16)v;   // transposed
      }
    }
  }
}

// ---------------------------------------------------------------------------
// Kernel 2: causal flash attention per (b, h, 64-query block)
// ---------------------------------------------------------------------------
__global__ __launch_bounds__(256) void attn_kernel(
    const __bf16* __restrict__ qws, const __bf16* __restrict__ kws,
    const __bf16* __restrict__ vws, __bf16* __restrict__ aws) {
  __shared__ __align__(16) __bf16 lq [64][72];   // [q-row][dh]
  __shared__ __align__(16) __bf16 lk [64][72];   // [key][dh]  (== K^T B-layout)
  __shared__ __align__(16) __bf16 lvT[64][72];   // [dh][key]
  __shared__ __align__(16) __bf16 lp [64][72];   // [q-row][key]
  __shared__ __align__(16) float  ls [64][65];   // scores f32
  __shared__ float lpm[64][4], lps[64][4];       // softmax partials
  __shared__ float lalpha[64];
  __shared__ float lrow[64];

  const int qblk = blockIdx.x, h = blockIdx.y, b = blockIdx.z;
  const int tid = threadIdx.x, lane = tid & 31, wv = tid >> 5;
  const int srow = tid & 63, sseg = tid >> 6;    // softmax: 4 segs x 16 cols
  const size_t head = ((size_t)b * H_ + h) * (size_t)N_ * DH_;

  // resident Q tile (bf16, pre-scaled by 1/8) via async LDS-direct copies
  {
    int idx = tid * 16;
    int r = idx >> 6, c = idx & 63;
    const __bf16* s = qws + head + ((size_t)(qblk * 64 + r)) * DH_ + c;
    g2l_b128(&lq[r][c], s);
    g2l_b128(&lq[r][c + 8], s + 8);
  }

  const int mtile = wv >> 1;         // 4 M-tiles
  const int nt0   = (wv & 1) * 2;    // 2 of 4 N-tiles per wave
  v8f o[2] = {};
  float mstate = -INFINITY, lstate = 0.f;
  async_wait();
  __syncthreads();

  // Q A-fragments are invariant over the key loop: hoist them
  const int arow = mtile * 16 + (lane & 15);
  const int hb   = (lane >> 4) * 8;
  const int kb   = (lane >> 4) * 16;
  v16bf aq[2];
  aq[0] = ldfrag(&lq[arow][hb],      &lq[arow][16 + hb]);
  aq[1] = ldfrag(&lq[arow][32 + hb], &lq[arow][48 + hb]);

  for (int j = 0; j <= qblk; ++j) {
    // stage K (row-major) and V (pre-transposed) via async LDS-direct copies
    {
      int idx = tid * 16;
      int r = idx >> 6, c = idx & 63;
      const __bf16* sk = kws + head + ((size_t)(j * 64 + r)) * DH_ + c;
      g2l_b128(&lk[r][c], sk);
      g2l_b128(&lk[r][c + 8], sk + 8);
      const __bf16* sv = vws + head + (size_t)r * N_ + (size_t)j * 64 + c;
      g2l_b128(&lvT[r][c], sv);
      g2l_b128(&lvT[r][c + 8], sv + 8);
    }
    async_wait();
    __syncthreads();

    // S = Q K^T (pre-scaled): preload 4 B-frags, then 4 chained WMMAs
    {
      v16bf bk[2][2];
      for (int t = 0; t < 2; ++t) {
        int col = (nt0 + t) * 16 + (lane & 15);   // key index
        bk[t][0] = ldfrag(&lk[col][hb],      &lk[col][16 + hb]);
        bk[t][1] = ldfrag(&lk[col][32 + hb], &lk[col][48 + hb]);
      }
      for (int t = 0; t < 2; ++t) {
        v8f s = {};
        s = __builtin_amdgcn_wmma_f32_16x16x32_bf16(
            false, aq[0], false, bk[t][0], (short)0, s, false, false);
        s = __builtin_amdgcn_wmma_f32_16x16x32_bf16(
            false, aq[1], false, bk[t][1], (short)0, s, false, false);
        int colb = (nt0 + t) * 16 + (lane & 15);
        int rb = mtile * 16 + ((lane >> 4) * 8);
        for (int r = 0; r < 8; ++r) ls[rb + r][colb] = s[r];
      }
    }
    __syncthreads();

    // online softmax, 4 threads per row (16 cols each)
    const int rowg = qblk * 64 + srow;
    const int cmax = rowg - j * 64;            // cols <= cmax are causal-valid
    const int c0 = sseg * 16;
    {
      float pm = -INFINITY;
      for (int c = 0; c < 16; ++c) {
        int cc = c0 + c;
        float v = (cc <= cmax) ? ls[srow][cc] : -INFINITY;
        pm = fmaxf(pm, v);
      }
      lpm[srow][sseg] = pm;
    }
    __syncthreads();
    float mx = fmaxf(fmaxf(lpm[srow][0], lpm[srow][1]),
                     fmaxf(lpm[srow][2], lpm[srow][3]));
    mx = fmaxf(mx, mstate);
    float alpha = __expf(mstate - mx);
    if (sseg == 0) lalpha[srow] = alpha;
    {
      float psum = 0.f;
      for (int c = 0; c < 16; c += 2) {
        int cc = c0 + c;
        float p0 = (cc     <= cmax) ? __expf(ls[srow][cc]     - mx) : 0.f;
        float p1 = (cc + 1 <= cmax) ? __expf(ls[srow][cc + 1] - mx) : 0.f;
        union { __bf16 h[2]; unsigned u; } pk;
        pk.h[0] = (__bf16)p0; pk.h[1] = (__bf16)p1;
        *(unsigned*)&lp[srow][cc] = pk.u;
        psum += p0 + p1;
      }
      lps[srow][sseg] = psum;
    }
    mstate = mx;
    __syncthreads();
    if (sseg == 0)
      lstate = lstate * alpha +
               (lps[srow][0] + lps[srow][1]) + (lps[srow][2] + lps[srow][3]);

    // O = O*alpha + P V : preload 6 frags, then chained WMMAs
    {
      v16bf ap[2], bv[2][2];
      ap[0] = ldfrag(&lp[arow][hb],      &lp[arow][16 + hb]);
      ap[1] = ldfrag(&lp[arow][32 + hb], &lp[arow][48 + hb]);
      for (int t = 0; t < 2; ++t) {
        int col = (nt0 + t) * 16 + (lane & 15);   // dh index
        bv[t][0] = ldfrag(&lvT[col][hb],      &lvT[col][16 + hb]);
        bv[t][1] = ldfrag(&lvT[col][32 + hb], &lvT[col][48 + hb]);
      }
      int rb = mtile * 16 + ((lane >> 4) * 8);
      for (int t = 0; t < 2; ++t)
        for (int r = 0; r < 8; ++r) o[t][r] *= lalpha[rb + r];
      for (int t = 0; t < 2; ++t) {
        o[t] = __builtin_amdgcn_wmma_f32_16x16x32_bf16(
            false, ap[0], false, bv[t][0], (short)0, o[t], false, false);
        o[t] = __builtin_amdgcn_wmma_f32_16x16x32_bf16(
            false, ap[1], false, bv[t][1], (short)0, o[t], false, false);
      }
    }
    __syncthreads();
  }

  if (tid < 64) lrow[tid] = 1.0f / lstate;
  __syncthreads();

  // bounce O through LDS (lp) for 128-bit coalesced global stores
  for (int t = 0; t < 2; ++t) {
    int colb = (nt0 + t) * 16 + (lane & 15);
    int rb = mtile * 16 + ((lane >> 4) * 8);
    for (int r = 0; r < 8; ++r)
      lp[rb + r][colb] = (__bf16)(o[t][r] * lrow[rb + r]);
  }
  __syncthreads();
  {
    int idx = tid * 16;
    int r = idx >> 6, c = idx & 63;
    uint4* dst = (uint4*)(aws + ((size_t)b * N_ + qblk * 64 + r) * D_ + h * DH_ + c);
    const uint4* s = (const uint4*)&lp[r][c];
    dst[0] = s[0]; dst[1] = s[1];
  }
}

// ---------------------------------------------------------------------------
// Kernel 3: output projection  out(4096x1024) = attn @ W_proj + b_proj (f32)
// ---------------------------------------------------------------------------
__global__ __launch_bounds__(256) void proj_kernel(
    const __bf16* __restrict__ aws, const float* __restrict__ Wp,
    const float* __restrict__ bp, float* __restrict__ out) {
  __shared__ __align__(16) __bf16 la [64][40];    // attn tile, row-major
  __shared__ __align__(16) __bf16 lbT[128][40];   // W_proj tile transposed [e][d]
  const int mblk = blockIdx.x, nblk = blockIdx.y;
  const int tid = threadIdx.x, lane = tid & 31, wv = tid >> 5;
  const int mtile = wv >> 1, nt0 = (wv & 1) * 4;
  v8f acc[4] = {};

  for (int kc = 0; kc < D_ / 32; ++kc) {
    // stage A tile 64x32 via async LDS-direct copies (bf16 scratch)
    {
      int idx = tid * 8;
      int r = idx >> 5, c = idx & 31;
      g2l_b128(&la[r][c],
               aws + ((size_t)(mblk * 64 + r)) * D_ + kc * 32 + c);
    }
    // stage W_proj tile 32x128 transposed -> lbT[e][d]
    {
      int idx = tid * 16;
      int r = idx >> 7, c = idx & 127;        // r: d-row, c: e-col
      const float* s = Wp + (size_t)(kc * 32 + r) * D_ + nblk * 128 + c;
      for (int i = 0; i < 16; i += 4) {
        float4 f = *(const float4*)(s + i);
        lbT[c + i    ][r] = (__bf16)f.x;
        lbT[c + i + 1][r] = (__bf16)f.y;
        lbT[c + i + 2][r] = (__bf16)f.z;
        lbT[c + i + 3][r] = (__bf16)f.w;
      }
    }
    if (kc + 1 < D_ / 32) {
      __builtin_prefetch(Wp + (size_t)((kc + 1) * 32 + (tid & 31)) * D_ + nblk * 128, 0, 3);
    }
    async_wait();
    __syncthreads();

    const int row = mtile * 16 + (lane & 15);
    const int hb  = (lane >> 4) * 8;
    const int kb  = (lane >> 4) * 16;
    v16bf a = ldfrag(&la[row][hb], &la[row][16 + hb]);
    v16bf bm[4];
    for (int t = 0; t < 4; ++t) {
      int col = (nt0 + t) * 16 + (lane & 15);
      bm[t] = ldfrag(&lbT[col][hb], &lbT[col][16 + hb]);
    }
    for (int t = 0; t < 4; ++t)
      acc[t] = __builtin_amdgcn_wmma_f32_16x16x32_bf16(
          false, a, false, bm[t], (short)0, acc[t], false, false);
    __syncthreads();
  }

  for (int t = 0; t < 4; ++t) {
    int col = (nt0 + t) * 16 + (lane & 15);
    int ng = nblk * 128 + col;
    float bias = bp[ng];
    for (int r = 0; r < 8; ++r) {
      int row = mtile * 16 + r + ((lane >> 4) * 8);
      out[((size_t)(mblk * 64 + row)) * D_ + ng] = acc[t][r] + bias;
    }
  }
}

// ---------------------------------------------------------------------------
extern "C" void kernel_launch(void* const* d_in, const int* in_sizes, int n_in,
                              void* d_out, int out_size, void* d_ws, size_t ws_size,
                              hipStream_t stream) {
  const float* x    = (const float*)d_in[0];
  const float* Wqkv = (const float*)d_in[1];
  const float* bqkv = (const float*)d_in[2];
  const float* Wp   = (const float*)d_in[3];
  const float* bp   = (const float*)d_in[4];
  float* out = (float*)d_out;

  char* ws = (char*)d_ws;
  const size_t per = (size_t)B_ * H_ * N_ * DH_ * sizeof(__bf16);  // 8 MB
  __bf16* kws = (__bf16*)(ws);
  __bf16* qws = (__bf16*)(ws + per);
  __bf16* vws = (__bf16*)(ws + 2 * per);          // [b][h][dh][n] (transposed)
  __bf16* aws = (__bf16*)(ws + 3 * per);          // B*N*D bf16

  dim3 blk(256);
  qkv_kernel<<<dim3(N_ / 64, H_, B_), blk, 0, stream>>>(x, Wqkv, bqkv, kws, qws, vws);
  attn_kernel<<<dim3(N_ / 64, H_, B_), blk, 0, stream>>>(qws, kws, vws, aws);
  proj_kernel<<<dim3((B_ * N_) / 64, D_ / 128), blk, 0, stream>>>(aws, Wp, bp, out);
}
